// Block_1236950581860
// MI455X (gfx1250) — matrix operand order repaired and checked by reading, not compiled, not run on hardware
//
#include <hip/hip_runtime.h>

// ---- problem sizes (compile-time) ----
#define B_   4
#define T_   2048
#define E_   1024
#define H_   16
#define HS_  64
#define BT_  (B_ * T_)
#define FF_  (4 * E_)

typedef __attribute__((ext_vector_type(16))) __bf16 v16bf;
typedef __attribute__((ext_vector_type(8)))  __bf16 v8bf;
typedef __attribute__((ext_vector_type(8)))  float  v8f;

union Frag16 { v16bf v; v8bf h[2]; };

__device__ __forceinline__ unsigned short f2bfu(float f) {
  unsigned int u = __float_as_uint(f);
  u += 0x7fffu + ((u >> 16) & 1u);         // round-to-nearest-even
  return (unsigned short)(u >> 16);
}

__device__ __forceinline__ v8f wmma_bf16(const Frag16& a, const Frag16& b, v8f c) {
  // D = A(16x32 bf16) * B(32x16 bf16) + C(16x16 f32)
  return __builtin_amdgcn_wmma_f32_16x16x32_bf16(
      /*neg_a=*/false, a.v, /*neg_b=*/false, b.v,
      /*c_mod=*/(short)0, c, /*reuse_a=*/false, /*reuse_b=*/false);
}

__device__ __forceinline__ float rmax16(float v) {
  v = fmaxf(v, __shfl_xor(v, 1, 32));
  v = fmaxf(v, __shfl_xor(v, 2, 32));
  v = fmaxf(v, __shfl_xor(v, 4, 32));
  v = fmaxf(v, __shfl_xor(v, 8, 32));
  return v;
}
__device__ __forceinline__ float rsum16(float v) {
  v += __shfl_xor(v, 1, 32);
  v += __shfl_xor(v, 2, 32);
  v += __shfl_xor(v, 4, 32);
  v += __shfl_xor(v, 8, 32);
  return v;
}

// ---------------- fp32 -> bf16 transpose: out[n][k] = bf16(in[k][n]) ----------------
// 32x32 LDS tile, 256 threads; grid (N/32, K/32).
__global__ __launch_bounds__(256)
void f2bfT_kernel(const float* __restrict__ in, unsigned short* __restrict__ out,
                  int K, int N) {
  __shared__ unsigned short t[32][33];
  const int k0 = blockIdx.y * 32, n0 = blockIdx.x * 32;
  const int tx = threadIdx.x & 31, ty = threadIdx.x >> 5;   // ty = 0..7
#pragma unroll
  for (int i = 0; i < 4; ++i) {
    int kk = ty + i * 8;
    t[kk][tx] = f2bfu(in[(size_t)(k0 + kk) * N + n0 + tx]);
  }
  __syncthreads();
#pragma unroll
  for (int i = 0; i < 4; ++i) {
    int nn = ty + i * 8;
    out[(size_t)(n0 + nn) * K + k0 + tx] = t[tx][nn];
  }
}

// ---------------- LayerNorm (weight only), output bf16 ----------------
// One block per row of E_=1024; 256 threads, 4 elems each.
__global__ __launch_bounds__(256)
void ln_kernel(const float* __restrict__ x, const float* __restrict__ w,
               unsigned short* __restrict__ out) {
  __shared__ float red[256];
  const int row = blockIdx.x;
  const int tid = threadIdx.x;
  const float* xr = x + (size_t)row * E_;
  float4 xv = *(const float4*)(xr + tid * 4);

  float s = xv.x + xv.y + xv.z + xv.w;
  red[tid] = s; __syncthreads();
  for (int off = 128; off > 0; off >>= 1) {
    if (tid < off) red[tid] += red[tid + off];
    __syncthreads();
  }
  const float mu = red[0] * (1.0f / E_);
  __syncthreads();

  float d0 = xv.x - mu, d1 = xv.y - mu, d2 = xv.z - mu, d3 = xv.w - mu;
  red[tid] = d0 * d0 + d1 * d1 + d2 * d2 + d3 * d3; __syncthreads();
  for (int off = 128; off > 0; off >>= 1) {
    if (tid < off) red[tid] += red[tid + off];
    __syncthreads();
  }
  const float rstd = rsqrtf(red[0] * (1.0f / E_) + 1e-5f);

  float4 wv = *(const float4*)(w + tid * 4);
  unsigned int p0 = (unsigned)f2bfu(d0 * rstd * wv.x) | ((unsigned)f2bfu(d1 * rstd * wv.y) << 16);
  unsigned int p1 = (unsigned)f2bfu(d2 * rstd * wv.z) | ((unsigned)f2bfu(d3 * rstd * wv.w) << 16);
  uint2 pk; pk.x = p0; pk.y = p1;
  *(uint2*)(out + (size_t)row * E_ + tid * 4) = pk;
}

// ---------------- Tiled bf16 WMMA GEMM ----------------
// C[M,N] = act(A[M,K] @ W[K,N] + bias) (+ res).
// A bf16 [M,K]; Wt = bf16 W-transposed [N,K]; bias/res f32.
// WG = 256 threads (8 waves); tile 128(M) x 64(N), K-chunk 64.
// Each wave: 16 rows x 64 cols (4 C-tiles, 8 WMMAs per K-chunk).
template<bool RELU>
__global__ __launch_bounds__(256)
void gemm_bf16_kernel(const unsigned short* __restrict__ A,
                      const unsigned short* __restrict__ Wt,
                      const float* __restrict__ bias,
                      const float* __restrict__ res,
                      float* __restrict__ outF,
                      unsigned short* __restrict__ outB,
                      int M, int N, int K) {
  constexpr int TK = 64;
  constexpr int LDA = TK + 8;                 // 72 (rows 16B-aligned)
  __shared__ unsigned short As[128][LDA];     // [m][k]
  __shared__ unsigned short Bs[64][LDA];      // [n][k]  (== W^T tile)

  const int tile_m = blockIdx.y * 128;
  const int tile_n = blockIdx.x * 64;
  const int tid  = threadIdx.x;
  const int wave = tid >> 5;
  const int lane = tid & 31;
  const int g    = lane >> 4;
  const int ln   = lane & 15;

  v8f acc[4];
#pragma unroll
  for (int t = 0; t < 4; ++t) acc[t] = v8f{0.f,0.f,0.f,0.f,0.f,0.f,0.f,0.f};

  for (int k0 = 0; k0 < K; k0 += TK) {
    // ---- stage A: 128 x 64, vector 8-elem (16B) chunks, 4 per thread ----
#pragma unroll
    for (int i = 0; i < 4; ++i) {
      int c = tid + i * 256;                 // 0..1023
      int r = c >> 3, cc = (c & 7) * 8;
      *(v8bf*)&As[r][cc] = *(const v8bf*)(A + (size_t)(tile_m + r) * K + k0 + cc);
    }
    // ---- stage W^T: 64 x 64, 2 chunks per thread ----
#pragma unroll
    for (int i = 0; i < 2; ++i) {
      int c = tid + i * 256;                 // 0..511
      int n = c >> 3, cc = (c & 7) * 8;
      *(v8bf*)&Bs[n][cc] = *(const v8bf*)(Wt + (size_t)(tile_n + n) * K + k0 + cc);
    }
    if (k0 + TK < K) {  // prefetch next chunks -> global_prefetch_b8
      __builtin_prefetch(A  + (size_t)(tile_m + (tid >> 1)) * K + k0 + TK + (tid & 1) * 32, 0, 0);
      __builtin_prefetch(Wt + (size_t)(tile_n + (tid & 63)) * K + k0 + TK + (tid >> 6) * 16, 0, 0);
    }
    __syncthreads();

    // ---- compute: 2 K-sub-chunks x 4 col tiles ----
#pragma unroll
    for (int c = 0; c < 2; ++c) {
      Frag16 a;
      const unsigned short* arow = &As[wave * 16 + ln][c * 32];
      a.h[0] = *(const v8bf*)(arow + g * 8);
      a.h[1] = *(const v8bf*)(arow + g * 8 + 16);
#pragma unroll
      for (int t = 0; t < 4; ++t) {
        Frag16 b;
        const unsigned short* brow = &Bs[t * 16 + ln][c * 32];
        b.h[0] = *(const v8bf*)(brow + g * 8);
        b.h[1] = *(const v8bf*)(brow + g * 8 + 16);
        acc[t] = wmma_bf16(a, b, acc[t]);
      }
    }
    __syncthreads();
  }

  // ---- epilogue: row = tile_m + wave*16 + g*8 + r ; col = tile_n + t*16 + ln ----
#pragma unroll
  for (int t = 0; t < 4; ++t) {
    const int col = tile_n + t * 16 + ln;
    const float bb = bias ? bias[col] : 0.0f;
#pragma unroll
    for (int r = 0; r < 8; ++r) {
      const int row = tile_m + wave * 16 + g * 8 + r;
      const size_t o = (size_t)row * N + col;
      float xv = acc[t][r] + bb;
      if (RELU) xv = fmaxf(xv, 0.0f);
      if (res)  xv += res[o];
      if (outF) outF[o] = xv;
      if (outB) outB[o] = f2bfu(xv);
    }
  }
}

// ---------------- Flash attention (causal), bf16 WMMA ----------------
// q,k,v,y are bf16 [BT, E] (head h lives at columns h*64..h*64+63).
// grid: (T/64, H, B); block 128 = 4 waves; each wave owns 16 query rows.
__global__ __launch_bounds__(128)
void attn_kernel(const unsigned short* __restrict__ q,
                 const unsigned short* __restrict__ k,
                 const unsigned short* __restrict__ v,
                 unsigned short* __restrict__ y) {
  constexpr int KB = 32;                         // key block
  __shared__ unsigned short Ks[KB][HS_ + 8];     // [key][d]     (B^T tile for QK^T)
  __shared__ unsigned short Vs[HS_][KB + 8];     // [d][key]     (B^T tile for P*V)
  __shared__ unsigned short Ps[4][16][KB];       // per-wave P relayout slab

  const int b   = blockIdx.z;
  const int h   = blockIdx.y;
  const int qb0 = blockIdx.x * 64;
  const int tid = threadIdx.x;
  const int wave = tid >> 5;
  const int lane = tid & 31;
  const int g  = lane >> 4;
  const int ln = lane & 15;
  const int qbase = qb0 + wave * 16;
  const size_t headcol = (size_t)h * HS_;

  // Q A-fragments for this wave's 16 rows (held in regs for whole loop)
  const unsigned short* qrow = q + ((size_t)(b * T_) + qbase + ln) * E_ + headcol;
  Frag16 qa0, qa1;
  qa0.h[0] = *(const v8bf*)(qrow + g * 8);
  qa0.h[1] = *(const v8bf*)(qrow + g * 8 + 16);
  qa1.h[0] = *(const v8bf*)(qrow + 32 + g * 8);
  qa1.h[1] = *(const v8bf*)(qrow + 32 + g * 8 + 16);

  v8f o[4];
#pragma unroll
  for (int t = 0; t < 4; ++t) o[t] = v8f{0.f,0.f,0.f,0.f,0.f,0.f,0.f,0.f};
  float mrow[8], lrow[8];
#pragma unroll
  for (int r = 0; r < 8; ++r) { mrow[r] = -1e30f; lrow[r] = 0.0f; }

  // reference scales by E^-0.5 = 1/32; fold log2(e) for exp2-domain softmax.
  const float sl2 = 0.03125f * 1.44269504088896f;
  const int kend = qb0 + 64;                     // causal upper bound for this block

  for (int kb = 0; kb < kend; kb += KB) {
    // ---- stage K block [32 keys][64 d] (row-major == B^T layout for QK^T) ----
    for (int i = tid; i < (KB * HS_) / 8; i += 128) {
      int rr = i >> 3, cc = (i & 7) * 8;
      *(v8bf*)&Ks[rr][cc] =
          *(const v8bf*)(k + ((size_t)(b * T_) + kb + rr) * E_ + headcol + cc);
    }
    // ---- stage V transposed: Vs[d][key] ----
    for (int i = tid; i < KB * HS_; i += 128) {
      int rr = i >> 6, cc = i & 63;   // rr = key, cc = d (coalesced read)
      Vs[cc][rr] = v[((size_t)(b * T_) + kb + rr) * E_ + headcol + cc];
    }
    __syncthreads();

    // ---- S = Q @ K^T : two 16x16 tiles (keys kb..kb+15, kb+16..kb+31) ----
    v8f s0 = {0.f,0.f,0.f,0.f,0.f,0.f,0.f,0.f};
    v8f s1 = {0.f,0.f,0.f,0.f,0.f,0.f,0.f,0.f};
    {
      Frag16 b0, b1;
      const unsigned short* kr0 = &Ks[ln][0];
      b0.h[0] = *(const v8bf*)(kr0 + g * 8);
      b0.h[1] = *(const v8bf*)(kr0 + g * 8 + 16);
      b1.h[0] = *(const v8bf*)(kr0 + 32 + g * 8);
      b1.h[1] = *(const v8bf*)(kr0 + 32 + g * 8 + 16);
      s0 = wmma_bf16(qa0, b0, s0);
      s0 = wmma_bf16(qa1, b1, s0);
      const unsigned short* kr1 = &Ks[16 + ln][0];
      b0.h[0] = *(const v8bf*)(kr1 + g * 8);
      b0.h[1] = *(const v8bf*)(kr1 + g * 8 + 16);
      b1.h[0] = *(const v8bf*)(kr1 + 32 + g * 8);
      b1.h[1] = *(const v8bf*)(kr1 + 32 + g * 8 + 16);
      s1 = wmma_bf16(qa0, b0, s1);
      s1 = wmma_bf16(qa1, b1, s1);
    }

    // ---- online softmax (per-row across 16 lanes of the half-wave group) ----
#pragma unroll
    for (int r = 0; r < 8; ++r) {
      const int qi = qbase + g * 8 + r;
      float a0 = s0[r] * sl2; if (kb + ln      > qi) a0 = -1e30f;
      float a1 = s1[r] * sl2; if (kb + 16 + ln > qi) a1 = -1e30f;
      float mx = rmax16(fmaxf(a0, a1));
      float mnew = fmaxf(mrow[r], mx);
      float corr = exp2f(mrow[r] - mnew);
      mrow[r] = mnew;
      float p0 = exp2f(a0 - mnew);
      float p1 = exp2f(a1 - mnew);
      lrow[r] = lrow[r] * corr + rsum16(p0 + p1);
#pragma unroll
      for (int t = 0; t < 4; ++t) o[t][r] *= corr;
      // park P (C-layout) into this wave's private LDS slab
      Ps[wave][g * 8 + r][ln]      = f2bfu(p0);
      Ps[wave][g * 8 + r][16 + ln] = f2bfu(p1);
    }

    // ---- reload P in A-fragment layout (same-wave LDS ops are in-order) ----
    Frag16 pa;
    const unsigned short* prow = &Ps[wave][ln][0];
    pa.h[0] = *(const v8bf*)(prow + g * 8);
    pa.h[1] = *(const v8bf*)(prow + g * 8 + 16);

    // ---- O += P @ V : four 16-col d-tiles ----
#pragma unroll
    for (int t = 0; t < 4; ++t) {
      Frag16 vb;
      const unsigned short* vrow = &Vs[t * 16 + ln][0];
      vb.h[0] = *(const v8bf*)(vrow + g * 8);
      vb.h[1] = *(const v8bf*)(vrow + g * 8 + 16);
      o[t] = wmma_bf16(pa, vb, o[t]);
    }
    __syncthreads();
  }

  // ---- normalize + write y (bf16) ----
#pragma unroll
  for (int r = 0; r < 8; ++r) {
    const float inv = 1.0f / lrow[r];
    const size_t base = ((size_t)(b * T_) + qbase + g * 8 + r) * E_ + headcol;
#pragma unroll
    for (int t = 0; t < 4; ++t) y[base + t * 16 + ln] = f2bfu(o[t][r] * inv);
  }
}

// ---------------- host-side orchestration ----------------
extern "C" void kernel_launch(void* const* d_in, const int* in_sizes, int n_in,
                              void* d_out, int out_size, void* d_ws, size_t ws_size,
                              hipStream_t stream) {
  const float* x    = (const float*)d_in[0];
  const float* Wq   = (const float*)d_in[1];
  const float* Wk   = (const float*)d_in[2];
  const float* Wv   = (const float*)d_in[3];
  const float* Wo   = (const float*)d_in[4];
  const float* bo   = (const float*)d_in[5];
  const float* ln1w = (const float*)d_in[6];
  const float* ln2w = (const float*)d_in[7];
  const float* W1   = (const float*)d_in[8];
  const float* b1   = (const float*)d_in[9];
  const float* W2   = (const float*)d_in[10];
  const float* b2   = (const float*)d_in[11];
  float* out = (float*)d_out;

  // workspace carve-up
  char* ws = (char*)d_ws;
  size_t off = 0;
  auto take = [&](size_t bytes) -> void* {
    void* p = ws + off;
    off += (bytes + 255) & ~(size_t)255;
    return p;
  };
  unsigned short* hb   = (unsigned short*)take((size_t)BT_ * E_ * 2);
  unsigned short* qb   = (unsigned short*)take((size_t)BT_ * E_ * 2);
  unsigned short* kb   = (unsigned short*)take((size_t)BT_ * E_ * 2);
  unsigned short* vb   = (unsigned short*)take((size_t)BT_ * E_ * 2);
  unsigned short* yb   = (unsigned short*)take((size_t)BT_ * E_ * 2);
  float*          xmid = (float*)take((size_t)BT_ * E_ * 4);
  unsigned short* h2b  = (unsigned short*)take((size_t)BT_ * E_ * 2);
  unsigned short* ffb  = (unsigned short*)take((size_t)BT_ * FF_ * 2);
  unsigned short* wqT  = (unsigned short*)take((size_t)E_ * E_ * 2);   // [N=E][K=E]
  unsigned short* wkT  = (unsigned short*)take((size_t)E_ * E_ * 2);
  unsigned short* wvT  = (unsigned short*)take((size_t)E_ * E_ * 2);
  unsigned short* woT  = (unsigned short*)take((size_t)E_ * E_ * 2);
  unsigned short* w1T  = (unsigned short*)take((size_t)E_ * FF_ * 2);  // [N=FF][K=E]
  unsigned short* w2T  = (unsigned short*)take((size_t)FF_ * E_ * 2);  // [N=E][K=FF]

  // 1) weight convert+transpose fp32[K,N] -> bf16[N,K]
  dim3 tEE(E_ / 32, E_ / 32);
  f2bfT_kernel<<<tEE, 256, 0, stream>>>(Wq, wqT, E_, E_);
  f2bfT_kernel<<<tEE, 256, 0, stream>>>(Wk, wkT, E_, E_);
  f2bfT_kernel<<<tEE, 256, 0, stream>>>(Wv, wvT, E_, E_);
  f2bfT_kernel<<<tEE, 256, 0, stream>>>(Wo, woT, E_, E_);
  f2bfT_kernel<<<dim3(FF_ / 32, E_ / 32), 256, 0, stream>>>(W1, w1T, E_, FF_);
  f2bfT_kernel<<<dim3(E_ / 32, FF_ / 32), 256, 0, stream>>>(W2, w2T, FF_, E_);

  // 2) h = LN(x, ln1_w)  (bf16)
  ln_kernel<<<BT_, 256, 0, stream>>>(x, ln1w, hb);

  // 3) q,k,v = h @ W{q,k,v}  (bf16 out)
  dim3 gEE(E_ / 64, BT_ / 128);
  gemm_bf16_kernel<false><<<gEE, 256, 0, stream>>>(hb, wqT, nullptr, nullptr, nullptr, qb, BT_, E_, E_);
  gemm_bf16_kernel<false><<<gEE, 256, 0, stream>>>(hb, wkT, nullptr, nullptr, nullptr, kb, BT_, E_, E_);
  gemm_bf16_kernel<false><<<gEE, 256, 0, stream>>>(hb, wvT, nullptr, nullptr, nullptr, vb, BT_, E_, E_);

  // 4) causal flash attention -> y (bf16, heads concatenated)
  attn_kernel<<<dim3(T_ / 64, H_, B_), 128, 0, stream>>>(qb, kb, vb, yb);

  // 5) x_mid = x + y @ Wo + bo  (f32)
  gemm_bf16_kernel<false><<<gEE, 256, 0, stream>>>(yb, woT, bo, x, xmid, nullptr, BT_, E_, E_);

  // 6) h2 = LN(x_mid, ln2_w)  (bf16)
  ln_kernel<<<BT_, 256, 0, stream>>>(xmid, ln2w, h2b);

  // 7) ff = relu(h2 @ W1 + b1)  (bf16)
  dim3 gFF(FF_ / 64, BT_ / 128);
  gemm_bf16_kernel<true><<<gFF, 256, 0, stream>>>(h2b, w1T, b1, nullptr, nullptr, ffb, BT_, FF_, E_);

  // 8) out = x_mid + ff @ W2 + b2  (f32)
  gemm_bf16_kernel<false><<<gEE, 256, 0, stream>>>(ffb, w2T, b2, xmid, out, nullptr, BT_, E_, FF_);
}
